// SparseCRB3d_28449863368848
// MI455X (gfx1250) — matrix operand order repaired and checked by reading, not compiled, not run on hardware
//
#include <hip/hip_runtime.h>

// ---------------------------------------------------------------------------
// Sparse 3D conv (rulebook gather-GEMM-scatter) + bias + ReLU + BatchNorm1d
// for MI455X (gfx1250, wave32, WMMA).
//
// Precision strategy: f32 emulated via bf16 hi/lo split (3x WMMA bf16 terms,
// f32 accumulate) -> ~f32 accuracy, uses V_WMMA_F32_16X16X32_BF16 which covers
// the full CIN=32 reduction per instruction.
// ---------------------------------------------------------------------------

typedef __attribute__((ext_vector_type(16))) __bf16 v16bf;
typedef __attribute__((ext_vector_type(8)))  float  v8f;

#define CIN   32
#define COUT  64
#define TILEP 16
#define WAVES 8
#define BN_EPS 1e-5f

__device__ __forceinline__ unsigned pack_bf2(__bf16 a, __bf16 b) {
    union { struct { __bf16 lo, hi; } s; unsigned u; } cv;
    cv.s.lo = a; cv.s.hi = b;
    return cv.u;
}

__device__ __forceinline__ void split_bf(float x, __bf16& hi, __bf16& lo) {
    hi = (__bf16)x;
    lo = (__bf16)(x - (float)hi);
}

// -------------------------------- zero fill --------------------------------
__global__ void zero_f(float* __restrict__ p, long n) {
    long i = (long)blockIdx.x * blockDim.x + threadIdx.x;
    long stride = (long)gridDim.x * blockDim.x;
    for (; i < n; i += stride) p[i] = 0.0f;
}

// --------------------------- main WMMA conv kernel -------------------------
// grid.x = K (kernel offsets), grid.y = tile groups (8 waves/block, 16 pairs/wave)
__global__ __launch_bounds__(256) void spconv_wmma(
    const float* __restrict__ features,   // [N, 32]
    const int*   __restrict__ in_idx,     // [K, P]
    const int*   __restrict__ out_idx,    // [K, P]
    const float* __restrict__ weight,     // [K, 32, 64]
    float*       __restrict__ acc,        // [N, 64] f32 accumulator
    int n_pairs, int tiles_per_k)
{
    // B-matrix staged in WMMA lane layout, bf16 hi/lo, packed 2-per-dword:
    // [nt][lane][j] ; lane<16: K={kb,kb+1}, kb=2j+8*(j>=4), N=nt*16+lane
    //                lane>=16: K={kb,kb+1}, kb=8+2j+8*(j>=4), N=nt*16+(lane-16)
    __shared__ __align__(16) unsigned sBhi[4][32][8];
    __shared__ __align__(16) unsigned sBlo[4][32][8];

    const int k   = blockIdx.x;
    const int tid = threadIdx.x;

    const float* W = weight + (long)k * (CIN * COUT);
    for (int e = tid; e < 4 * 32 * 8; e += 256) {
        int j    = e & 7;
        int ln   = (e >> 3) & 31;
        int nt   = e >> 8;
        int n    = nt * 16 + (ln & 15);
        int kb   = ((ln >> 4) << 3) + 2 * j + ((j >> 2) << 3);
        float w0 = W[(long)kb * COUT + n];
        float w1 = W[(long)(kb + 1) * COUT + n];
        __bf16 h0, l0, h1, l1;
        split_bf(w0, h0, l0);
        split_bf(w1, h1, l1);
        sBhi[nt][ln][j] = pack_bf2(h0, h1);
        sBlo[nt][ln][j] = pack_bf2(l0, l1);
    }
    __syncthreads();

    const int wave  = tid >> 5;
    const int lane  = tid & 31;
    const int laneM = lane & 15;
    const int half  = lane >> 4;

    const int tile = blockIdx.y * WAVES + wave;
    if (tile >= tiles_per_k) return;
    const int p0 = tile * TILEP;

    // ---- gather A tile in ISA A-matrix layout ----
    // lane<16 : row = in_idx[p0+lane],   K elems {0..7} U {16..23}
    // lane>=16: row = in_idx[p0+lane-16],K elems {8..15} U {24..31}
    const int  p      = p0 + laneM;
    const bool pvalid = (p < n_pairs);
    const int  row    = pvalid ? in_idx[(long)k * n_pairs + p] : 0;
    const float* fr   = features + (long)row * CIN;
    const int cb      = half * 8;

    float4 f0, f1, f2, f3;
    if (pvalid) {
        f0 = *(const float4*)(fr + cb);
        f1 = *(const float4*)(fr + cb + 4);
        f2 = *(const float4*)(fr + 16 + cb);
        f3 = *(const float4*)(fr + 16 + cb + 4);
    } else {
        f0 = f1 = f2 = f3 = make_float4(0.f, 0.f, 0.f, 0.f);
    }

    float av[16] = { f0.x, f0.y, f0.z, f0.w,  f1.x, f1.y, f1.z, f1.w,
                     f2.x, f2.y, f2.z, f2.w,  f3.x, f3.y, f3.z, f3.w };
    v16bf ahi, alo;
#pragma unroll
    for (int e = 0; e < 16; ++e) {
        __bf16 h, l;
        split_bf(av[e], h, l);
        ahi[e] = h;
        alo[e] = l;
    }

    const int* oix = out_idx + (long)k * n_pairs;

#pragma unroll
    for (int nt = 0; nt < 4; ++nt) {
        union { v16bf v; uint4 u[2]; } bh, bl;
        bh.u[0] = *(const uint4*)&sBhi[nt][lane][0];
        bh.u[1] = *(const uint4*)&sBhi[nt][lane][4];
        bl.u[0] = *(const uint4*)&sBlo[nt][lane][0];
        bl.u[1] = *(const uint4*)&sBlo[nt][lane][4];

        v8f c = {};
        // f32-emulating bf16x3: hi*hi + hi*lo + lo*hi (lo*lo ~ 2^-32, dropped)
        c = __builtin_amdgcn_wmma_f32_16x16x32_bf16(false, ahi, false, bh.v,
                                                    (short)0, c, false, false);
        c = __builtin_amdgcn_wmma_f32_16x16x32_bf16(false, ahi, false, bl.v,
                                                    (short)0, c, false, false);
        c = __builtin_amdgcn_wmma_f32_16x16x32_bf16(false, alo, false, bh.v,
                                                    (short)0, c, false, false);

        // ---- scatter D tile (C/D layout: vgpr r -> M = r + 8*half, N = laneM)
#pragma unroll
        for (int r = 0; r < 8; ++r) {
            const int M  = r + half * 8;
            const int pm = p0 + M;
            if (pm < n_pairs) {
                const int orow = oix[pm];
                atomicAdd(&acc[(long)orow * COUT + nt * 16 + laneM], c[r]);
            }
        }
    }
}

// ------------------- BN stats pass 1: per-chunk partial sums ----------------
// block b reduces rows [b*rpb, b*rpb+rpb) over all 64 channels (coalesced).
__global__ __launch_bounds__(256) void bn_partial(
    const float* __restrict__ acc, const float* __restrict__ bias,
    float* __restrict__ partial, int nrows, int rpb)
{
    __shared__ float ls[256];
    __shared__ float lss[256];
    const int b  = blockIdx.x;
    const int t  = threadIdx.x;
    const int ch = t & 63;
    const int rs = t >> 6;
    const int r0 = b * rpb;
    const int r1 = min(nrows, r0 + rpb);
    const float bi = bias[ch];

    float s = 0.f, ss = 0.f;
    for (int r = r0 + rs; r < r1; r += 4) {
        float v = acc[(long)r * COUT + ch] + bi;
        v = v > 0.f ? v : 0.f;   // ReLU BEFORE BatchNorm (matches reference)
        s  += v;
        ss += v * v;
    }
    ls[t] = s; lss[t] = ss;
    __syncthreads();
    if (t < 64) {
        float S  = ls[t]  + ls[t + 64]  + ls[t + 128]  + ls[t + 192];
        float SS = lss[t] + lss[t + 64] + lss[t + 128] + lss[t + 192];
        partial[(long)b * 128 + t]      = S;
        partial[(long)b * 128 + 64 + t] = SS;
    }
}

// ------------------- BN stats pass 2: combine (deterministic) ---------------
__global__ void bn_finalize(const float* __restrict__ partial,
                            float* __restrict__ stats, int nblocks, int nrows)
{
    const int ch = threadIdx.x;   // 64 threads
    float s = 0.f, ss = 0.f;
    for (int b = 0; b < nblocks; ++b) {
        s  += partial[(long)b * 128 + ch];
        ss += partial[(long)b * 128 + 64 + ch];
    }
    const float mean = s / (float)nrows;
    const float var  = ss / (float)nrows - mean * mean;  // population var
    stats[ch]      = mean;
    stats[64 + ch] = var;
}

// --------------------------- apply bias/ReLU/BN -----------------------------
__global__ void bn_apply(float* __restrict__ out, const float* __restrict__ bias,
                         const float* __restrict__ gamma,
                         const float* __restrict__ beta,
                         const float* __restrict__ stats, long n)
{
    long i = (long)blockIdx.x * blockDim.x + threadIdx.x;
    long stride = (long)gridDim.x * blockDim.x;
    for (; i < n; i += stride) {
        const int ch = (int)(i & 63);
        float v = out[i] + bias[ch];
        v = v > 0.f ? v : 0.f;
        const float m  = stats[ch];
        const float vr = stats[64 + ch];
        out[i] = (v - m) * rsqrtf(vr + BN_EPS) * gamma[ch] + beta[ch];
    }
}

// ---------------------------------------------------------------------------
extern "C" void kernel_launch(void* const* d_in, const int* in_sizes, int n_in,
                              void* d_out, int out_size, void* d_ws, size_t ws_size,
                              hipStream_t stream)
{
    const float* features = (const float*)d_in[0];
    const int*   in_idx   = (const int*)d_in[1];
    const int*   out_idx  = (const int*)d_in[2];
    const float* weight   = (const float*)d_in[3];
    const float* bias     = (const float*)d_in[4];
    const float* gamma    = (const float*)d_in[5];
    const float* beta     = (const float*)d_in[6];
    float* out = (float*)d_out;

    const int K = in_sizes[3] / (CIN * COUT);    // 27
    const int P = in_sizes[1] / K;               // 60000
    const int N = in_sizes[0] / CIN;             // 150000
    const long total = (long)N * COUT;

    // 1) zero accumulator (graph replays don't re-poison d_out)
    zero_f<<<1024, 256, 0, stream>>>(out, total);

    // 2) gather-WMMA-scatter
    const int tiles = (P + TILEP - 1) / TILEP;
    dim3 g1(K, (tiles + WAVES - 1) / WAVES);
    spconv_wmma<<<g1, 256, 0, stream>>>(features, in_idx, out_idx, weight,
                                        out, P, tiles);

    // 3) deterministic BN stats (ReLU(x+bias) over active voxels)
    const int NB  = 256;
    const int rpb = (N + NB - 1) / NB;
    float* partial = (float*)d_ws;          // NB*128 floats = 128 KB
    float* stats   = partial + (long)NB * 128;  // 128 floats
    bn_partial<<<NB, 256, 0, stream>>>(out, bias, partial, N, rpb);
    bn_finalize<<<1, 64, 0, stream>>>(partial, stats, NB, N);

    // 4) normalize in place
    int blocks3 = (int)((total + 255) / 256);
    if (blocks3 > 4096) blocks3 = 4096;
    bn_apply<<<blocks3, 256, 0, stream>>>(out, bias, gamma, beta, stats, total);
}
